// Encoder2_41721312313627
// MI455X (gfx1250) — compile-verified, matrix operand order checked by reading, not run on hardware
//
#include <hip/hip_runtime.h>

// ---------------- constants ----------------
constexpr int   Bq     = 4;
constexpr int   NSEG   = 8;
constexpr int   SLEN   = 512;
constexpr int   DVAR   = 32;
constexpr int   HH     = 256;      // H
constexpr int   WK     = 16;
constexpr int   NHEADS = 8;
constexpr int   NSTATE = 64;
constexpr int   TCq    = SLEN - WK + 1;   // 497
constexpr int   HD     = HH / NHEADS;     // 32
constexpr int   BN     = Bq * NSEG;       // 32
constexpr long long ROWS = (long long)BN * TCq;   // 15904 (divisible by 32)

typedef __attribute__((ext_vector_type(16))) _Float16 v16h;
typedef __attribute__((ext_vector_type(8)))  float    v8f;

enum { ACT_NONE = 0, ACT_RELU = 1 };
enum { BIAS_COL = 1, BIAS_ROW = 2 };

// ---------------- WMMA helpers ----------------
__device__ __forceinline__ v8f wmma32(v16h a, v16h b, v8f c) {
  // D = A(16x32 f16) x B(32x16 f16) + C(16x16 f32)
  return __builtin_amdgcn_wmma_f32_16x16x32_f16(false, a, false, b, (short)0, c,
                                                false, false);
}

// Fragment layout [ISA 7.12.2]: element e of lane (r=lane&15, hi=lane>>4) holds
// M(row)=row0+r, K = k0 + (e<8 ? 8*hi+e : 16+8*hi+(e-8)).

// Unguarded row-major fragment load (two contiguous b128 per lane).
__device__ __forceinline__ v16h load_rm_frag_u(const _Float16* __restrict__ base,
                                               int row0, int ld, int k0, int lane) {
  int r = lane & 15, hi = lane >> 4;
  v16h f;
  const _Float16* p = base + (long long)(row0 + r) * ld + k0 + 8 * hi;
#pragma unroll
  for (int e = 0; e < 8; ++e) f[e] = p[e];
#pragma unroll
  for (int e = 0; e < 8; ++e) f[8 + e] = p[16 + e];
  return f;
}

// Row-guarded variant (edge tiles).
__device__ __forceinline__ v16h load_rm_frag(const _Float16* __restrict__ base,
                                             int row0, int ld, int k0, int rmax,
                                             int lane) {
  int r = lane & 15, hi = lane >> 4;
  v16h f;
  int row = row0 + r;
  if (row < rmax) {
    const _Float16* p = base + (long long)row * ld + k0 + 8 * hi;
#pragma unroll
    for (int e = 0; e < 8; ++e) f[e] = p[e];
#pragma unroll
    for (int e = 0; e < 8; ++e) f[8 + e] = p[16 + e];
  } else {
#pragma unroll
    for (int e = 0; e < 16; ++e) f[e] = (_Float16)0;
  }
  return f;
}

// B fragment where the source is K-major (B[k][n], row stride ld): used for P@V.
template <bool GUARD>
__device__ __forceinline__ v16h load_km_frag(const _Float16* __restrict__ base,
                                             int n0, int ld, int k0, int kmax,
                                             int lane) {
  int r = lane & 15, hi = lane >> 4;
  v16h f;
#pragma unroll
  for (int e = 0; e < 16; ++e) {
    int kk = (e < 8) ? (8 * hi + e) : (16 + 8 * hi + (e - 8));
    int k  = k0 + kk;
    if (GUARD)
      f[e] = (k < kmax) ? base[(long long)k * ld + n0 + r] : (_Float16)0;
    else
      f[e] = base[(long long)k * ld + n0 + r];
  }
  return f;
}

// ---------------- generic f16 GEMM: C = A * Bt^T (+bias, act) ----------------
// A: MxK row-major (lda). Bt: NxK row-major (ldb) -> out[m][n]=sum_k A[m,k]*Bt[n,k]
// Block: 128 threads = 4 waves arranged 2x2; wave tile 64(M) x 64(N);
// block tile 128x128. 16 WMMAs per K-step per wave.
template <typename OT, int ACT, int BIASMODE>
__global__ void __launch_bounds__(128)
gemm_f16_wmma(const _Float16* __restrict__ A, const _Float16* __restrict__ Bt,
              OT* __restrict__ C, const float* __restrict__ bias,
              int M, int N, int K, int lda, int ldb, int ldc,
              long long sA, long long sB, long long sC) {
  int lane = threadIdx.x & 31;
  int wave = threadIdx.x >> 5;
  long long bz = blockIdx.z;
  const _Float16* Ab = A + bz * sA;
  const _Float16* Bb = Bt + bz * sB;
  OT* Cb = C + bz * sC;
  int m0 = blockIdx.y * 128 + (wave >> 1) * 64;
  int n0 = blockIdx.x * 128 + (wave & 1) * 64;
  int r = lane & 15, hi = lane >> 4;

  v8f acc[4][4];
#pragma unroll
  for (int i = 0; i < 4; ++i)
#pragma unroll
    for (int j = 0; j < 4; ++j)
#pragma unroll
      for (int a = 0; a < 8; ++a) acc[i][j][a] = 0.0f;

  bool full = (m0 + 64 <= M) && (n0 + 64 <= N);

  if (full) {
    for (int k0 = 0; k0 < K; k0 += 32) {
      v16h af[4];
#pragma unroll
      for (int i = 0; i < 4; ++i)
        af[i] = load_rm_frag_u(Ab, m0 + 16 * i, lda, k0, lane);
#pragma unroll
      for (int j = 0; j < 4; ++j) {
        v16h bf = load_rm_frag_u(Bb, n0 + 16 * j, ldb, k0, lane);
#pragma unroll
        for (int i = 0; i < 4; ++i) acc[i][j] = wmma32(af[i], bf, acc[i][j]);
      }
    }
#pragma unroll
    for (int i = 0; i < 4; ++i)
#pragma unroll
      for (int j = 0; j < 4; ++j)
#pragma unroll
        for (int a = 0; a < 8; ++a) {
          int row = m0 + 16 * i + a + 8 * hi;
          int col = n0 + 16 * j + r;
          float v = acc[i][j][a];
          if (BIASMODE == BIAS_COL) v += bias[col];
          else                      v += bias[row];
          if (ACT == ACT_RELU) v = fmaxf(v, 0.0f);
          Cb[(long long)row * ldc + col] = (OT)v;
        }
  } else {
    for (int k0 = 0; k0 < K; k0 += 32) {
      v16h af[4];
#pragma unroll
      for (int i = 0; i < 4; ++i)
        af[i] = load_rm_frag(Ab, m0 + 16 * i, lda, k0, M, lane);
#pragma unroll
      for (int j = 0; j < 4; ++j) {
        v16h bf = load_rm_frag(Bb, n0 + 16 * j, ldb, k0, N, lane);
#pragma unroll
        for (int i = 0; i < 4; ++i) acc[i][j] = wmma32(af[i], bf, acc[i][j]);
      }
    }
#pragma unroll
    for (int i = 0; i < 4; ++i)
#pragma unroll
      for (int j = 0; j < 4; ++j)
#pragma unroll
        for (int a = 0; a < 8; ++a) {
          int row = m0 + 16 * i + a + 8 * hi;
          int col = n0 + 16 * j + r;
          if (row < M && col < N) {
            float v = acc[i][j][a];
            if (BIASMODE == BIAS_COL) v += bias[col];
            else                      v += bias[row];
            if (ACT == ACT_RELU) v = fmaxf(v, 0.0f);
            Cb[(long long)row * ldc + col] = (OT)v;
          }
        }
  }
}

// ---------------- fused GLU GEMM: out = (A@Wg[0:H].T+b0) * sigmoid(A@Wg[H:2H].T+b1)
// N dimension (HH=256) is always full; only M can be ragged.
__global__ void __launch_bounds__(128)
glu_gemm_wmma(const _Float16* __restrict__ A, const _Float16* __restrict__ Wg,
              float* __restrict__ Out, const float* __restrict__ bias, int M) {
  int lane = threadIdx.x & 31;
  int wave = threadIdx.x >> 5;
  int m0 = blockIdx.y * 64 + (wave >> 1) * 32;
  int n0 = blockIdx.x * 64 + (wave & 1) * 32;
  int r = lane & 15, hi = lane >> 4;

  v8f av[2][2], ag[2][2];
#pragma unroll
  for (int i = 0; i < 2; ++i)
#pragma unroll
    for (int j = 0; j < 2; ++j)
#pragma unroll
      for (int a = 0; a < 8; ++a) { av[i][j][a] = 0.0f; ag[i][j][a] = 0.0f; }

  bool full = (m0 + 32 <= M);
  if (full) {
    for (int k0 = 0; k0 < HH; k0 += 32) {
      v16h a0 = load_rm_frag_u(A, m0, HH, k0, lane);
      v16h a1 = load_rm_frag_u(A, m0 + 16, HH, k0, lane);
#pragma unroll
      for (int j = 0; j < 2; ++j) {
        v16h bv = load_rm_frag_u(Wg, n0 + 16 * j, HH, k0, lane);
        v16h bg = load_rm_frag_u(Wg, HH + n0 + 16 * j, HH, k0, lane);
        av[0][j] = wmma32(a0, bv, av[0][j]);
        av[1][j] = wmma32(a1, bv, av[1][j]);
        ag[0][j] = wmma32(a0, bg, ag[0][j]);
        ag[1][j] = wmma32(a1, bg, ag[1][j]);
      }
    }
  } else {
    for (int k0 = 0; k0 < HH; k0 += 32) {
      v16h a0 = load_rm_frag(A, m0, HH, k0, M, lane);
      v16h a1 = load_rm_frag(A, m0 + 16, HH, k0, M, lane);
#pragma unroll
      for (int j = 0; j < 2; ++j) {
        v16h bv = load_rm_frag_u(Wg, n0 + 16 * j, HH, k0, lane);
        v16h bg = load_rm_frag_u(Wg, HH + n0 + 16 * j, HH, k0, lane);
        av[0][j] = wmma32(a0, bv, av[0][j]);
        av[1][j] = wmma32(a1, bv, av[1][j]);
        ag[0][j] = wmma32(a0, bg, ag[0][j]);
        ag[1][j] = wmma32(a1, bg, ag[1][j]);
      }
    }
  }

#pragma unroll
  for (int i = 0; i < 2; ++i)
#pragma unroll
    for (int j = 0; j < 2; ++j)
#pragma unroll
      for (int a = 0; a < 8; ++a) {
        int row = m0 + 16 * i + a + 8 * hi;
        int col = n0 + 16 * j + r;
        if (full || row < M) {
          float val  = av[i][j][a] + bias[col];
          float gate = ag[i][j][a] + bias[HH + col];
          float sig  = 1.0f / (1.0f + __expf(-gate));
          Out[(long long)row * HH + col] = val * sig;
        }
      }
}

// ---------------- flash attention (per wave: 16 q-rows, online softmax) --------
__global__ void __launch_bounds__(128)
attn_flash_wmma(const _Float16* __restrict__ Q, const _Float16* __restrict__ Km,
                const _Float16* __restrict__ V, _Float16* __restrict__ O) {
  __shared__ _Float16 lp[4][16][40];  // per-wave 16x32 P tile (+pad)
  int lane = threadIdx.x & 31;
  int wave = threadIdx.x >> 5;
  int b = blockIdx.z, head = blockIdx.y;
  int row0 = (blockIdx.x * 4 + wave) * 16;
  long long base = (long long)b * TCq * HH + head * HD;
  const _Float16* q = Q + base;
  const _Float16* k = Km + base;
  const _Float16* v = V + base;
  int r = lane & 15, hi = lane >> 4;
  const float sc = 0.17677669529663687f;  // 1/sqrt(32)

  v16h aq = load_rm_frag(q, row0, HH, 0, TCq, lane);

  v8f o0, o1;
  float mrun[8], lrun[8];
#pragma unroll
  for (int a = 0; a < 8; ++a) { o0[a] = 0.f; o1[a] = 0.f; mrun[a] = -1e30f; lrun[a] = 0.f; }

  for (int s0 = 0; s0 < TCq; s0 += 32) {
    bool fullc = (s0 + 32 <= TCq);  // uniform
    v16h bk0, bk1;
    if (fullc) {
      bk0 = load_rm_frag_u(k, s0, HH, 0, lane);
      bk1 = load_rm_frag_u(k, s0 + 16, HH, 0, lane);
    } else {
      bk0 = load_rm_frag(k, s0, HH, 0, TCq, lane);
      bk1 = load_rm_frag(k, s0 + 16, HH, 0, TCq, lane);
    }
    v8f z;
#pragma unroll
    for (int a = 0; a < 8; ++a) z[a] = 0.f;
    v8f S0 = wmma32(aq, bk0, z);
    v8f S1 = wmma32(aq, bk1, z);

    bool ok0 = fullc || ((s0 + r) < TCq);
    bool ok1 = fullc || ((s0 + 16 + r) < TCq);
#pragma unroll
    for (int a = 0; a < 8; ++a) {
      float s0v = ok0 ? S0[a] * sc : -1e30f;
      float s1v = ok1 ? S1[a] * sc : -1e30f;
      float cm = fmaxf(s0v, s1v);
      cm = fmaxf(cm, __shfl_xor(cm, 1));
      cm = fmaxf(cm, __shfl_xor(cm, 2));
      cm = fmaxf(cm, __shfl_xor(cm, 4));
      cm = fmaxf(cm, __shfl_xor(cm, 8));
      float mn    = fmaxf(mrun[a], cm);
      float scale = __expf(mrun[a] - mn);
      float p0 = __expf(s0v - mn);
      float p1 = __expf(s1v - mn);
      float rs = p0 + p1;
      rs += __shfl_xor(rs, 1);
      rs += __shfl_xor(rs, 2);
      rs += __shfl_xor(rs, 4);
      rs += __shfl_xor(rs, 8);
      lrun[a] = lrun[a] * scale + rs;
      mrun[a] = mn;
      o0[a] *= scale;
      o1[a] *= scale;
      int prow = a + 8 * hi;
      lp[wave][prow][r]      = (_Float16)p0;
      lp[wave][prow][16 + r] = (_Float16)p1;
    }
    asm volatile("s_wait_dscnt 0" ::: "memory");
    v16h ap = load_rm_frag_u(&lp[wave][0][0], 0, 40, 0, lane);
    v16h bv0, bv1;
    if (fullc) {
      bv0 = load_km_frag<false>(v, 0,  HH, s0, TCq, lane);
      bv1 = load_km_frag<false>(v, 16, HH, s0, TCq, lane);
    } else {
      bv0 = load_km_frag<true>(v, 0,  HH, s0, TCq, lane);
      bv1 = load_km_frag<true>(v, 16, HH, s0, TCq, lane);
    }
    o0 = wmma32(ap, bv0, o0);
    o1 = wmma32(ap, bv1, o1);
  }

#pragma unroll
  for (int a = 0; a < 8; ++a) {
    float inv = 1.0f / lrun[a];
    int row = row0 + a + 8 * hi;
    if (row < TCq) {
      O[base + (long long)row * HH + r]      = (_Float16)(o0[a] * inv);
      O[base + (long long)row * HH + 16 + r] = (_Float16)(o1[a] * inv);
    }
  }
}

// ---------------- S4D kernel build: Kt[l*H + h] ----------------
__global__ void __launch_bounds__(128)
s4_kernel_build(const float* __restrict__ log_dt, const float* __restrict__ log_A_re,
                const float* __restrict__ A_im, const float* __restrict__ B_re,
                const float* __restrict__ B_im, const float* __restrict__ C_re,
                const float* __restrict__ C_im, float* __restrict__ Kt) {
  int h = blockIdx.x;
  int t = threadIdx.x;
  __shared__ float cre[NSTATE], cim[NSTATE], dre[NSTATE], dimg[NSTATE];
  if (t < NSTATE) {
    int n = t;
    float dt  = __expf(log_dt[h]);
    float Are = -__expf(log_A_re[h * NSTATE + n]);
    float Aim = A_im[h * NSTATE + n];
    float dr = dt * Are, di = dt * Aim;
    float er = __expf(dr);
    float s, c;
    __sincosf(di, &s, &c);
    float em1r = er * c - 1.0f, em1i = er * s;
    float den = Are * Are + Aim * Aim;
    float dBr = (em1r * Are + em1i * Aim) / den;
    float dBi = (em1i * Are - em1r * Aim) / den;
    float Br = B_re[h * NSTATE + n], Bi = B_im[h * NSTATE + n];
    float tr = dBr * Br - dBi * Bi, ti = dBr * Bi + dBi * Br;
    float Cr = C_re[h * NSTATE + n], Ci = C_im[h * NSTATE + n];
    cre[n] = Cr * tr - Ci * ti;
    cim[n] = Cr * ti + Ci * tr;
    dre[n] = dr;
    dimg[n] = di;
  }
  __syncthreads();
  for (int l = t; l < TCq; l += 128) {
    float acc = 0.0f;
    float fl = (float)l;
    for (int n = 0; n < NSTATE; ++n) {
      float e = __expf(dre[n] * fl);
      float s, c;
      __sincosf(dimg[n] * fl, &s, &c);
      acc += cre[n] * e * c - cim[n] * e * s;
    }
    Kt[(long long)l * HH + h] = 2.0f * acc;
  }
}

// ---------------- S4 causal conv + skip + gelu (tanh approx) -> f16 ------------
__global__ void __launch_bounds__(256)
s4_conv_gelu(const float* __restrict__ U, const float* __restrict__ Kt,
             const float* __restrict__ Dv, _Float16* __restrict__ Yg) {
  int l = blockIdx.x, b = blockIdx.y, h = threadIdx.x;
  const float* ub = U + (long long)b * TCq * HH + h;
  float acc = Dv[h] * ub[(long long)l * HH];
  for (int j = 0; j <= l; ++j)
    acc += Kt[(long long)j * HH + h] * ub[(long long)(l - j) * HH];
  float x  = acc;
  float t  = tanhf(0.7978845608028654f * (x + 0.044715f * x * x * x));
  float g  = 0.5f * x * (1.0f + t);
  Yg[((long long)b * TCq + l) * HH + h] = (_Float16)g;
}

// ---------------- LayerNorm(X+Y) ----------------
template <bool W16>
__global__ void __launch_bounds__(256)
ln_residual(const float* __restrict__ X, const float* __restrict__ Y,
            const float* __restrict__ g, const float* __restrict__ bb,
            float* __restrict__ Out32, _Float16* __restrict__ Out16) {
  long long row = blockIdx.x;
  int h = threadIdx.x;
  __shared__ float s1[256], s2[256];
  float v = X[row * HH + h] + Y[row * HH + h];
  s1[h] = v;
  s2[h] = v * v;
  __syncthreads();
  for (int st = 128; st > 0; st >>= 1) {
    if (h < st) { s1[h] += s1[h + st]; s2[h] += s2[h + st]; }
    __syncthreads();
  }
  float mean = s1[0] * (1.0f / HH);
  float var  = s2[0] * (1.0f / HH) - mean * mean;
  float inv  = rsqrtf(var + 1e-5f);
  float o = (v - mean) * inv * g[h] + bb[h];
  Out32[row * HH + h] = o;
  if (W16) Out16[row * HH + h] = (_Float16)o;
}

// ---------------- final z[b,h] = sum_t out2[b,t,h]*lw[t] + lb ----------------
__global__ void __launch_bounds__(256)
final_reduce(const float* __restrict__ Out2, const float* __restrict__ lw,
             const float* __restrict__ lb, float* __restrict__ Z) {
  int b = blockIdx.x, h = threadIdx.x;
  const float* p = Out2 + (long long)b * TCq * HH + h;
  float acc = 0.0f;
  for (int t = 0; t < TCq; ++t) acc += p[(long long)t * HH] * lw[t];
  Z[b * HH + h] = acc + lb[0];
}

// ---------------- f32 -> f16 convert ----------------
__global__ void cvt_f32_f16(const float* __restrict__ in, _Float16* __restrict__ out,
                            long long n) {
  long long i = (long long)blockIdx.x * blockDim.x + threadIdx.x;
  if (i < n) out[i] = (_Float16)in[i];
}

// ================= host side =================
extern "C" void kernel_launch(void* const* d_in, const int* in_sizes, int n_in,
                              void* d_out, int out_size, void* d_ws, size_t ws_size,
                              hipStream_t stream) {
  const float* x       = (const float*)d_in[0];
  const float* conv1_w = (const float*)d_in[1];
  const float* conv1_b = (const float*)d_in[2];
  const float* wq = (const float*)d_in[3];
  const float* bq = (const float*)d_in[4];
  const float* wk = (const float*)d_in[5];
  const float* bk = (const float*)d_in[6];
  const float* wv = (const float*)d_in[7];
  const float* bv = (const float*)d_in[8];
  const float* wo = (const float*)d_in[9];
  const float* bo = (const float*)d_in[10];
  const float* log_dt   = (const float*)d_in[11];
  const float* log_A_re = (const float*)d_in[12];
  const float* A_im     = (const float*)d_in[13];
  const float* B_re = (const float*)d_in[14];
  const float* B_im = (const float*)d_in[15];
  const float* C_re = (const float*)d_in[16];
  const float* C_im = (const float*)d_in[17];
  const float* Dv   = (const float*)d_in[18];
  const float* w_glu = (const float*)d_in[19];
  const float* b_glu = (const float*)d_in[20];
  const float* ln1_g = (const float*)d_in[21];
  const float* ln1_b = (const float*)d_in[22];
  const float* ln2_g = (const float*)d_in[23];
  const float* ln2_b = (const float*)d_in[24];
  const float* ffn1_w = (const float*)d_in[25];
  const float* ffn1_b = (const float*)d_in[26];
  const float* ffn2_w = (const float*)d_in[27];
  const float* ffn2_b = (const float*)d_in[28];
  const float* lin_w  = (const float*)d_in[29];
  const float* lin_b  = (const float*)d_in[30];
  (void)in_sizes; (void)n_in; (void)out_size; (void)ws_size;

  // ---- workspace carve (~84 MB) ----
  char* ws = (char*)d_ws;
  size_t off = 0;
  auto take = [&](size_t bytes) -> void* {
    void* p = ws + off;
    off += (bytes + 255) & ~(size_t)255;
    return p;
  };
  const long long N_X   = (long long)BN * SLEN * DVAR;  // 524288
  const long long N_CW  = (long long)HH * WK * DVAR;    // 131072
  const long long N_W   = (long long)HH * HH;           // 65536
  const long long N_GLU = (long long)2 * HH * HH;       // 131072
  const long long ACT   = ROWS * HH;                    // 4,071,424

  _Float16* x16    = (_Float16*)take(N_X * 2);
  _Float16* cw16   = (_Float16*)take(N_CW * 2);
  _Float16* wq16   = (_Float16*)take(N_W * 2);
  _Float16* wk16   = (_Float16*)take(N_W * 2);
  _Float16* wv16   = (_Float16*)take(N_W * 2);
  _Float16* wo16   = (_Float16*)take(N_W * 2);
  _Float16* wglu16 = (_Float16*)take(N_GLU * 2);
  _Float16* f1_16  = (_Float16*)take(N_W * 2);
  _Float16* f2_16  = (_Float16*)take(N_W * 2);
  float*    Kt     = (float*)take((long long)TCq * HH * 4);
  _Float16* inp16  = (_Float16*)take(ACT * 2);  // conv out -> later reused as attn O
  _Float16* q16    = (_Float16*)take(ACT * 2);  // later reused as gelu(y) f16
  _Float16* k16    = (_Float16*)take(ACT * 2);  // later reused as x1 f16
  _Float16* v16    = (_Float16*)take(ACT * 2);  // later reused as ffn-mid f16
  float*    att    = (float*)take(ACT * 4);     // later reused as out2
  float*    x1     = (float*)take(ACT * 4);
  float*    tmp32  = (float*)take(ACT * 4);     // s4y then ffn y2

  auto cvt = [&](const float* src, _Float16* dst, long long n) {
    cvt_f32_f16<<<dim3((unsigned)((n + 255) / 256)), dim3(256), 0, stream>>>(src, dst, n);
  };
  cvt(x, x16, N_X);
  cvt(conv1_w, cw16, N_CW);
  cvt(wq, wq16, N_W);
  cvt(wk, wk16, N_W);
  cvt(wv, wv16, N_W);
  cvt(wo, wo16, N_W);
  cvt(w_glu, wglu16, N_GLU);
  cvt(ffn1_w, f1_16, N_W);
  cvt(ffn2_w, f2_16, N_W);

  // S4D convolution kernel K(h,l), stored transposed as Kt[l*H+h]
  s4_kernel_build<<<dim3(HH), dim3(128), 0, stream>>>(log_dt, log_A_re, A_im, B_re,
                                                      B_im, C_re, C_im, Kt);

  // Conv1 as implicit GEMM: out[b][h][t] (raw memory == inp[b][t][h] after reshape)
  gemm_f16_wmma<_Float16, ACT_RELU, BIAS_ROW>
      <<<dim3((TCq + 127) / 128, (HH + 127) / 128, BN), dim3(128), 0, stream>>>(
          cw16, x16, inp16, conv1_b, HH, TCq, WK * DVAR, WK * DVAR, DVAR, TCq,
          0LL, (long long)SLEN * DVAR, (long long)HH * TCq);

  // QKV projections (one big GEMM each over all 15904 rows)
  dim3 gq((HH + 127) / 128, (unsigned)((ROWS + 127) / 128), 1);
  gemm_f16_wmma<_Float16, ACT_NONE, BIAS_COL><<<gq, dim3(128), 0, stream>>>(
      inp16, wq16, q16, bq, (int)ROWS, HH, HH, HH, HH, HH, 0LL, 0LL, 0LL);
  gemm_f16_wmma<_Float16, ACT_NONE, BIAS_COL><<<gq, dim3(128), 0, stream>>>(
      inp16, wk16, k16, bk, (int)ROWS, HH, HH, HH, HH, HH, 0LL, 0LL, 0LL);
  gemm_f16_wmma<_Float16, ACT_NONE, BIAS_COL><<<gq, dim3(128), 0, stream>>>(
      inp16, wv16, v16, bv, (int)ROWS, HH, HH, HH, HH, HH, 0LL, 0LL, 0LL);

  // Flash attention; O written (f16) over the conv-out buffer.
  _Float16* o16 = inp16;
  attn_flash_wmma<<<dim3((TCq + 63) / 64, NHEADS, BN), dim3(128), 0, stream>>>(
      q16, k16, v16, o16);

  // Output projection -> att (f32)
  gemm_f16_wmma<float, ACT_NONE, BIAS_COL><<<gq, dim3(128), 0, stream>>>(
      o16, wo16, att, bo, (int)ROWS, HH, HH, HH, HH, HH, 0LL, 0LL, 0LL);

  // S4: causal conv + D skip + gelu -> f16 (reuse q16)
  _Float16* yg16 = q16;
  s4_conv_gelu<<<dim3(TCq, BN), dim3(256), 0, stream>>>(att, Kt, Dv, yg16);

  // Fused GLU GEMM -> s4y (tmp32)
  glu_gemm_wmma<<<dim3((HH + 63) / 64, (unsigned)((ROWS + 63) / 64)), dim3(128), 0,
                  stream>>>(yg16, wglu16, tmp32, b_glu, (int)ROWS);

  // x1 = LN(att + s4y)  (f32 + f16 copy in k16)
  _Float16* x1_16 = k16;
  ln_residual<true><<<dim3((unsigned)ROWS), dim3(256), 0, stream>>>(att, tmp32, ln1_g,
                                                                    ln1_b, x1, x1_16);

  // FFN: mid = relu(x1@ffn1^T+b) f16 (reuse v16); y2 = mid@ffn2^T+b f32 (tmp32)
  _Float16* mid16 = v16;
  gemm_f16_wmma<_Float16, ACT_RELU, BIAS_COL><<<gq, dim3(128), 0, stream>>>(
      x1_16, f1_16, mid16, ffn1_b, (int)ROWS, HH, HH, HH, HH, HH, 0LL, 0LL, 0LL);
  gemm_f16_wmma<float, ACT_NONE, BIAS_COL><<<gq, dim3(128), 0, stream>>>(
      mid16, f2_16, tmp32, ffn2_b, (int)ROWS, HH, HH, HH, HH, HH, 0LL, 0LL, 0LL);

  // out2 = LN(x1 + y2) -> reuse att buffer (f32 only)
  float* out2 = att;
  ln_residual<false><<<dim3((unsigned)ROWS), dim3(256), 0, stream>>>(
      x1, tmp32, ln2_g, ln2_b, out2, (_Float16*)nullptr);

  // z[b,h] = sum_t out2[b,t,h]*lin_w[t] + lin_b
  final_reduce<<<dim3(BN), dim3(256), 0, stream>>>(out2, lin_w, lin_b, (float*)d_out);
}
